// PointnetPP_82729660055784
// MI455X (gfx1250) — matrix operand order repaired and checked by reading, not compiled, not run on hardware
//
#include <hip/hip_runtime.h>

typedef __bf16 bf16;
typedef __bf16 v16bf __attribute__((ext_vector_type(16)));
typedef __bf16 v8bf  __attribute__((ext_vector_type(8)));
typedef float  v8f   __attribute__((ext_vector_type(8)));

constexpr int    BB   = 2;
constexpr int    NPTS = 8192;
constexpr int    KNN  = 128;
constexpr int    S1   = 4096;
constexpr float  R2   = 0.16f;                 // RADIUS^2
constexpr size_t M1   = (size_t)BB * S1 * KNN; // 1048576 rows (SA1 MLP)
constexpr size_t M2   = (size_t)BB * S1;       // 8192 rows (SA2 MLP)
constexpr int    KMAX = 160;                   // max padded K
constexpr int    WPAD = 8;                     // LDS pitch pad (bank-conflict break)

// ---------------------------------------------------------------------------
// FPS: one block per batch, serial over NS1 samples, block argmax each step.
// ---------------------------------------------------------------------------
__global__ __launch_bounds__(1024)
void fps_kernel(const float* __restrict__ pos, float* __restrict__ sampled)
{
    const int b   = blockIdx.x;
    const int tid = threadIdx.x;
    __shared__ float dmin[NPTS];
    __shared__ float wv[32];
    __shared__ int   wi[32];
    __shared__ int   s_last;

    const float* P = pos + (size_t)b * NPTS * 3;
    for (int j = tid; j < NPTS; j += 1024) dmin[j] = 1e10f;
    if (tid == 0) s_last = 0;
    __syncthreads();

    for (int it = 0; it < S1; ++it) {
        const int last = s_last;
        const float px = P[last * 3 + 0];
        const float py = P[last * 3 + 1];
        const float pz = P[last * 3 + 2];
        if (tid < 3) sampled[((size_t)b * S1 + it) * 3 + tid] = P[last * 3 + tid];

        float best = -1.f; int bidx = 0;
        for (int j = tid; j < NPTS; j += 1024) {
            const float dx = P[j * 3 + 0] - px;
            const float dy = P[j * 3 + 1] - py;
            const float dz = P[j * 3 + 2] - pz;
            const float d  = dx * dx + dy * dy + dz * dz;
            const float dm = fminf(dmin[j], d);
            dmin[j] = dm;
            if (dm > best || (dm == best && j < bidx)) { best = dm; bidx = j; }
        }
        #pragma unroll
        for (int off = 16; off > 0; off >>= 1) {
            const float ov = __shfl_down(best, off, 32);
            const int   oi = __shfl_down(bidx, off, 32);
            if (ov > best || (ov == best && oi < bidx)) { best = ov; bidx = oi; }
        }
        if ((tid & 31) == 0) { wv[tid >> 5] = best; wi[tid >> 5] = bidx; }
        __syncthreads();
        if (tid < 32) {
            best = wv[tid]; bidx = wi[tid];
            #pragma unroll
            for (int off = 16; off > 0; off >>= 1) {
                const float ov = __shfl_down(best, off, 32);
                const int   oi = __shfl_down(bidx, off, 32);
                if (ov > best || (ov == best && oi < bidx)) { best = ov; bidx = oi; }
            }
            if (tid == 0) s_last = bidx;
        }
        __syncthreads();
    }
}

// ---------------------------------------------------------------------------
// kNN: one block per query; distances in LDS; 3-level 256-bin radix select of
// the 128 smallest (f32 >= 0 is order-isomorphic to its uint bits).
// ---------------------------------------------------------------------------
__global__ __launch_bounds__(256)
void knn_kernel(const float* __restrict__ pos, const float* __restrict__ sampled,
                int* __restrict__ topk_idx, float* __restrict__ topk_d2)
{
    const int q   = blockIdx.x;     // b*S1 + s
    const int b   = q / S1;
    const int tid = threadIdx.x;
    __shared__ float    d2[NPTS];
    __shared__ unsigned hist[256];
    __shared__ unsigned s_t, s_base, s_cA, s_cB;

    const float* P  = pos + (size_t)b * NPTS * 3;
    const float  qx = sampled[(size_t)q * 3 + 0];
    const float  qy = sampled[(size_t)q * 3 + 1];
    const float  qz = sampled[(size_t)q * 3 + 2];

    for (int j = tid; j < NPTS; j += 256) {
        const float dx = P[j * 3 + 0] - qx;
        const float dy = P[j * 3 + 1] - qy;
        const float dz = P[j * 3 + 2] - qz;
        d2[j] = dx * dx + dy * dy + dz * dz;
    }
    // ---- pass 1: bits[31:24] ----
    hist[tid] = 0; __syncthreads();
    for (int j = tid; j < NPTS; j += 256)
        atomicAdd(&hist[__float_as_uint(d2[j]) >> 24], 1u);
    __syncthreads();
    if (tid == 0) {
        unsigned cum = 0; int t = 0;
        for (t = 0; t < 256; ++t) { if (cum + hist[t] >= 128u) break; cum += hist[t]; }
        s_t = (unsigned)t; s_base = cum;
    }
    __syncthreads();
    const unsigned t1 = s_t, base1 = s_base;
    // ---- pass 2: bits[23:16] among top-byte == t1 ----
    __syncthreads();
    hist[tid] = 0; __syncthreads();
    for (int j = tid; j < NPTS; j += 256) {
        const unsigned bits = __float_as_uint(d2[j]);
        if ((bits >> 24) == t1) atomicAdd(&hist[(bits >> 16) & 255u], 1u);
    }
    __syncthreads();
    if (tid == 0) {
        unsigned cum = base1; int t = 0;
        for (t = 0; t < 256; ++t) { if (cum + hist[t] >= 128u) break; cum += hist[t]; }
        s_t = (unsigned)t; s_base = cum;
    }
    __syncthreads();
    const unsigned t2 = s_t, base2 = s_base;
    // ---- pass 3: bits[15:8] among 16-bit prefix == (t1,t2) ----
    __syncthreads();
    hist[tid] = 0; __syncthreads();
    const unsigned pref16 = (t1 << 8) | t2;
    for (int j = tid; j < NPTS; j += 256) {
        const unsigned bits = __float_as_uint(d2[j]);
        if ((bits >> 16) == pref16) atomicAdd(&hist[(bits >> 8) & 255u], 1u);
    }
    __syncthreads();
    if (tid == 0) {
        unsigned cum = base2; int t = 0;
        for (t = 0; t < 256; ++t) { if (cum + hist[t] >= 128u) break; cum += hist[t]; }
        s_t = (unsigned)t; s_base = cum; s_cA = 0; s_cB = 0;
    }
    __syncthreads();
    const unsigned thr   = (pref16 << 8) | s_t;  // 24-bit prefix threshold
    const unsigned base3 = s_base;
    const unsigned rem   = 128u - base3;
    int*   oi = topk_idx + (size_t)q * KNN;
    float* od = topk_d2  + (size_t)q * KNN;
    for (int j = tid; j < NPTS; j += 256) {
        const unsigned key = __float_as_uint(d2[j]) >> 8;
        if (key < thr) {
            const unsigned slot = atomicAdd(&s_cA, 1u);
            oi[slot] = j; od[slot] = d2[j];
        } else if (key == thr) {
            const unsigned t = atomicAdd(&s_cB, 1u);
            if (t < rem) { oi[base3 + t] = j; od[base3 + t] = d2[j]; }
        }
    }
}

// ---------------------------------------------------------------------------
// Pack f32 weights [Cout,Cin] -> bf16 [Cout,Kpad] (zero pad).
// ---------------------------------------------------------------------------
__global__ __launch_bounds__(256)
void pack_weights(const float* __restrict__ W, bf16* __restrict__ Wb,
                  int cin, int kpad, int total)
{
    const int i = blockIdx.x * 256 + threadIdx.x;
    if (i >= total) return;
    const int o = i / kpad, k = i % kpad;
    Wb[i] = (k < cin) ? (bf16)W[o * cin + k] : (bf16)0.f;
}

__global__ __launch_bounds__(256)
void zero_f32(float* __restrict__ p, int n)
{
    const int i = blockIdx.x * 256 + threadIdx.x;
    if (i < n) p[i] = 0.f;
}

// ---------------------------------------------------------------------------
// Build grouped features [M1 x 32] bf16: {pos[idx]-q (3), x[idx] (3), 0-pad}.
// ---------------------------------------------------------------------------
__global__ __launch_bounds__(256)
void build_grouped(const float* __restrict__ x, const float* __restrict__ pos,
                   const float* __restrict__ sampled, const int* __restrict__ topk,
                   bf16* __restrict__ G)
{
    const size_t m = (size_t)blockIdx.x * 256 + threadIdx.x;
    if (m >= M1) return;
    const int s = (int)((m / KNN) % S1);
    const int b = (int)(m / ((size_t)KNN * S1));
    const int j = topk[m];
    const float* pj = pos + ((size_t)b * NPTS + j) * 3;
    const float* xj = x   + ((size_t)b * NPTS + j) * 3;
    const float* qq = sampled + ((size_t)b * S1 + s) * 3;
    bf16* g = G + m * 32;
    g[0] = (bf16)(pj[0] - qq[0]);
    g[1] = (bf16)(pj[1] - qq[1]);
    g[2] = (bf16)(pj[2] - qq[2]);
    g[3] = (bf16)xj[0];
    g[4] = (bf16)xj[1];
    g[5] = (bf16)xj[2];
    #pragma unroll
    for (int c = 6; c < 32; ++c) g[c] = (bf16)0.f;
}

// ---------------------------------------------------------------------------
// WMMA GEMM: Y[M,N] = relu_bn(A)[M,Kd] @ W[N,Kd]^T + bias, bf16 in/out,
// f32 accumulate; per-channel sum/sumsq for training-mode BN.
//  - 64-col weight tile staged in LDS once per block via async copy
//    (global_load_async_to_lds_b128 + s_wait_asynccnt), pitch padded to
//    break bank conflicts.
//  - Each wave owns 32 rows x 64 cols: 2 A fragments, 4 B fragments (from
//    LDS), 8 v_wmma_f32_16x16x32_bf16 per k-step.
//  - Previous layer's BN+ReLU folded into the A load.
// ---------------------------------------------------------------------------
__global__ __launch_bounds__(256)
void gemm_bn_wmma(const bf16* __restrict__ A, int lda,
                  const bf16* __restrict__ W, int ldw,
                  const float* __restrict__ bias,
                  const float* __restrict__ bnScale,
                  const float* __restrict__ bnShift,
                  bf16* __restrict__ Y, int ldy,
                  float* __restrict__ cSum, float* __restrict__ cSqs,
                  int N, int Kd)
{
    __shared__ float lsum[1024];
    __shared__ float lsqs[1024];
    __shared__ bf16  sW[64 * (KMAX + WPAD)];   // 64 cols x Kd (pitch-padded)
    for (int c = threadIdx.x; c < N; c += 256) { lsum[c] = 0.f; lsqs[c] = 0.f; }

    const int lane  = threadIdx.x & 31;
    const int wv    = threadIdx.x >> 5;
    const int h     = lane >> 4;      // lane half (K split for A, row split for D)
    const int ml    = lane & 15;      // A row within tile / output column
    const int pitch = Kd + WPAD;
    const size_t row0  = ((size_t)blockIdx.x * 8 + wv) * 32;
    const bf16*  arow0 = A + (row0 + ml) * (size_t)lda;
    const bf16*  arow1 = A + (row0 + 16 + ml) * (size_t)lda;
    const unsigned sWbase = (unsigned)(size_t)&sW[0];
    const int kd8 = Kd >> 3;          // 16B units per weight row

    for (int col0 = 0; col0 < N; col0 += 64) {
        __syncthreads();  // previous chunk's readers done before re-staging
        // ---- async-stage 64 x Kd weight tile into LDS ----
        for (int u = threadIdx.x; u < 64 * kd8; u += 256) {
            const int r    = u / kd8;
            const int koff = (u - r * kd8) * 8;
            const bf16* gp = W + (size_t)(col0 + r) * ldw + koff;
            const unsigned lb = sWbase + (unsigned)((r * pitch + koff) * 2);
            asm volatile("global_load_async_to_lds_b128 %0, %1, off"
                         :: "v"(lb), "v"(gp) : "memory");
        }
        asm volatile("s_wait_asynccnt 0" ::: "memory");
        __syncthreads();

        v8f acc[8] = {};
        for (int k0 = 0; k0 < Kd; k0 += 32) {
            // A fragments (two 16-row tiles), BN+ReLU of previous layer folded in
            const bf16* ap0 = arow0 + k0 + h * 8;
            const bf16* ap1 = arow1 + k0 + h * 8;
            if (k0 + 32 < Kd) { __builtin_prefetch(ap0 + 32, 0, 3); __builtin_prefetch(ap1 + 32, 0, 3); }
            const v8bf a0lo = *(const v8bf*)ap0;
            const v8bf a0hi = *(const v8bf*)(ap0 + 16);
            const v8bf a1lo = *(const v8bf*)ap1;
            const v8bf a1hi = *(const v8bf*)(ap1 + 16);
            v16bf a0, a1;
            if (bnScale) {
                #pragma unroll
                for (int i = 0; i < 8; ++i) {
                    const int kA = k0 + h * 8 + i;
                    const float scl = bnScale[kA],      shl = bnShift[kA];
                    const float sch = bnScale[kA + 16], shh = bnShift[kA + 16];
                    a0[i]     = (bf16)fmaxf((float)a0lo[i] * scl + shl, 0.f);
                    a0[i + 8] = (bf16)fmaxf((float)a0hi[i] * sch + shh, 0.f);
                    a1[i]     = (bf16)fmaxf((float)a1lo[i] * scl + shl, 0.f);
                    a1[i + 8] = (bf16)fmaxf((float)a1hi[i] * sch + shh, 0.f);
                }
            } else {
                #pragma unroll
                for (int i = 0; i < 8; ++i) {
                    a0[i] = a0lo[i]; a0[i + 8] = a0hi[i];
                    a1[i] = a1lo[i]; a1[i + 8] = a1hi[i];
                }
            }
            #pragma unroll
            for (int t = 0; t < 4; ++t) {
                // B fragment from LDS: column n = t*16+ml, K = k0 + h*8 .. +7 and k0 + 16 + h*8 .. +7
                const bf16* bp = &sW[(t * 16 + ml) * pitch + k0 + h * 8];
                const v8bf blo = *(const v8bf*)bp;
                const v8bf bhi = *(const v8bf*)(bp + 16);
                v16bf bfrag;
                #pragma unroll
                for (int i = 0; i < 8; ++i) { bfrag[i] = blo[i]; bfrag[i + 8] = bhi[i]; }
                acc[t]     = __builtin_amdgcn_wmma_f32_16x16x32_bf16(
                                 false, a0, false, bfrag, (short)0, acc[t],     false, false);
                acc[t + 4] = __builtin_amdgcn_wmma_f32_16x16x32_bf16(
                                 false, a1, false, bfrag, (short)0, acc[t + 4], false, false);
            }
        }
        // ---- epilogue: bias, BN stats, bf16 store ----
        #pragma unroll
        for (int rr = 0; rr < 2; ++rr) {
            const size_t rbase = row0 + rr * 16 + h * 8;
            #pragma unroll
            for (int t = 0; t < 4; ++t) {
                v8f& ac = acc[rr * 4 + t];
                const int   n  = col0 + t * 16 + ml;
                const float bv = bias[n];
                float s = 0.f, ss = 0.f;
                #pragma unroll
                for (int j = 0; j < 8; ++j) {
                    const float v = ac[j] + bv;
                    ac[j] = v;
                    s += v; ss += v * v;
                }
                s  += __shfl_xor(s, 16, 32);
                ss += __shfl_xor(ss, 16, 32);
                if (h == 0) { atomicAdd(&lsum[n], s); atomicAdd(&lsqs[n], ss); }
                bf16* yp = Y + rbase * (size_t)ldy + n;
                #pragma unroll
                for (int j = 0; j < 8; ++j) yp[(size_t)j * ldy] = (bf16)ac[j];
            }
        }
    }
    __syncthreads();
    for (int c = threadIdx.x; c < N; c += 256) {
        atomicAdd(&cSum[c], lsum[c]);
        atomicAdd(&cSqs[c], lsqs[c]);
    }
}

// ---------------------------------------------------------------------------
// BN finalize: per-channel scale/shift from accumulated sum/sumsq.
// ---------------------------------------------------------------------------
__global__ __launch_bounds__(256)
void finalize_bn(const float* __restrict__ sum, const float* __restrict__ sumsq,
                 const float* __restrict__ gamma, const float* __restrict__ beta,
                 float* __restrict__ scale, float* __restrict__ shift,
                 int N, float invM)
{
    const int c = blockIdx.x * 256 + threadIdx.x;
    if (c >= N) return;
    const float mu  = sum[c] * invM;
    const float var = fmaxf(sumsq[c] * invM - mu * mu, 0.f);
    const float sc  = gamma[c] * rsqrtf(var + 1e-5f);
    scale[c] = sc;
    shift[c] = beta[c] - mu * sc;
}

// ---------------------------------------------------------------------------
// SA1 epilogue: BN+ReLU, radius mask (d2 > R^2 -> -1e8), max over K=128.
// Writes x1 [M2,128] and transposed output rt_pp_feat [B,128,S1].
// ---------------------------------------------------------------------------
__global__ __launch_bounds__(128)
void maxpool_sa1(const bf16* __restrict__ Y3, const float* __restrict__ scale,
                 const float* __restrict__ shift, const float* __restrict__ topk_d2,
                 float* __restrict__ x1, float* __restrict__ out_rt)
{
    const int q = blockIdx.x;         // b*S1 + s
    const int c = threadIdx.x;        // 0..127
    const int b = q / S1, s = q % S1;
    const bf16*  y  = Y3 + (size_t)q * KNN * 128 + c;
    const float* dd = topk_d2 + (size_t)q * KNN;
    const float  sc = scale[c], sh = shift[c];
    float m = -1e30f;
    for (int k = 0; k < KNN; ++k) {
        float v = fmaxf((float)y[(size_t)k * 128] * sc + sh, 0.f);
        if (dd[k] > R2) v -= 1e8f;
        m = fmaxf(m, v);
    }
    x1[(size_t)q * 128 + c] = m;
    out_rt[((size_t)b * 128 + c) * S1 + s] = m;
}

// ---------------------------------------------------------------------------
// Build SA2 input [M2 x 160] bf16: {sampled_pos (3), x1 (128), 0-pad}.
// ---------------------------------------------------------------------------
__global__ __launch_bounds__(256)
void build_g2(const float* __restrict__ sampled, const float* __restrict__ x1,
              bf16* __restrict__ G)
{
    const size_t i = (size_t)blockIdx.x * 256 + threadIdx.x;
    if (i >= M2 * 160) return;
    const size_t r = i / 160;
    const int    c = (int)(i % 160);
    float v = 0.f;
    if (c < 3)        v = sampled[r * 3 + c];
    else if (c < 131) v = x1[r * 128 + (c - 3)];
    G[i] = (bf16)v;
}

// ---------------------------------------------------------------------------
// SA2 epilogue: BN+ReLU then max over all S1 points -> global_x [B,1024].
// ---------------------------------------------------------------------------
__global__ __launch_bounds__(256)
void global_max(const bf16* __restrict__ Y6, const float* __restrict__ scale,
                const float* __restrict__ shift, float* __restrict__ outg)
{
    const int i = blockIdx.x * 256 + threadIdx.x;
    if (i >= BB * 1024) return;
    const int b = i >> 10, c = i & 1023;
    const float sc = scale[c], sh = shift[c];
    const bf16* y = Y6 + (size_t)b * S1 * 1024 + c;
    float m = -1e30f;
    for (int s = 0; s < S1; ++s)
        m = fmaxf(m, fmaxf((float)y[(size_t)s * 1024] * sc + sh, 0.f));
    outg[i] = m;
}

// ---------------------------------------------------------------------------
// Host launcher
// ---------------------------------------------------------------------------
extern "C" void kernel_launch(void* const* d_in, const int* in_sizes, int n_in,
                              void* d_out, int out_size, void* d_ws, size_t ws_size,
                              hipStream_t stream)
{
    (void)in_sizes; (void)n_in; (void)out_size; (void)ws_size;

    const float* x   = (const float*)d_in[0];
    const float* pos = (const float*)d_in[1];
    const float* Wp[6];  const float* bp[6];  const float* gp[6];  const float* ep[6];
    for (int l = 0; l < 6; ++l) {
        Wp[l] = (const float*)d_in[2 + 4 * l + 0];
        bp[l] = (const float*)d_in[2 + 4 * l + 1];
        gp[l] = (const float*)d_in[2 + 4 * l + 2];
        ep[l] = (const float*)d_in[2 + 4 * l + 3];
    }
    static const int cin[6]  = {6, 64, 64, 131, 128, 128};
    static const int kpad[6] = {32, 64, 64, 160, 128, 128};
    static const int cout[6] = {64, 64, 128, 128, 128, 1024};

    // ---- workspace carve-up ----
    char*  ws  = (char*)d_ws;
    size_t off = 0;
    auto alloc = [&](size_t bytes) -> char* {
        char* p = ws + off;
        off = (off + bytes + 255) & ~(size_t)255;
        return p;
    };
    float* sampled = (float*)alloc((size_t)BB * S1 * 3 * 4);
    float* topkd2  = (float*)alloc(M1 * 4);
    float* cStats  = (float*)alloc(2048 * 4);          // [sum(1024) | sumsq(1024)]
    float* cSum    = cStats;
    float* cSqs    = cStats + 1024;
    float* scaleL[6]; float* shiftL[6];
    for (int l = 0; l < 6; ++l) {
        scaleL[l] = (float*)alloc(1024 * 4);
        shiftL[l] = (float*)alloc(1024 * 4);
    }
    bf16* Wb[6];
    for (int l = 0; l < 6; ++l) Wb[l] = (bf16*)alloc((size_t)cout[l] * kpad[l] * 2);
    float* x1   = (float*)alloc(M2 * 128 * 4);
    bf16*  g2   = (bf16*)alloc(M2 * 160 * 2);
    bf16*  Y4   = (bf16*)alloc(M2 * 128 * 2);
    bf16*  Y5   = (bf16*)alloc(M2 * 128 * 2);
    bf16*  Y6   = (bf16*)alloc(M2 * 1024 * 2);
    bf16*  bufP = (bf16*)alloc(M1 * 64 * 2);   // grouped(32) then Y2(64)
    bf16*  bufQ = (bf16*)alloc(M1 * 128 * 2);  // Y1(64) then Y3(128)

    // ---- output carve-up (flat, return order) ----
    float* out_rt = (float*)d_out;                         // [B,128,S1]
    float* out_gx = out_rt + (size_t)BB * 128 * S1;        // [B,1024]
    int*   out_ti = (int*)(out_gx + (size_t)BB * 1024);    // [B,S1,K] int bits

    const float invM1 = 1.f / (float)M1;
    const float invM2 = 1.f / (float)M2;

    // 1) FPS -> sampled_pos
    fps_kernel<<<BB, 1024, 0, stream>>>(pos, sampled);
    // 2) kNN -> topk_idx (into output) + d^2
    knn_kernel<<<BB * S1, 256, 0, stream>>>(pos, sampled, out_ti, topkd2);
    // 3) pack weights to padded bf16
    for (int l = 0; l < 6; ++l) {
        const int tot = cout[l] * kpad[l];
        pack_weights<<<(tot + 255) / 256, 256, 0, stream>>>(Wp[l], Wb[l], cin[l], kpad[l], tot);
    }
    // 4) gather grouped features -> bufP [M1 x 32]
    build_grouped<<<(int)(M1 / 256), 256, 0, stream>>>(x, pos, sampled, out_ti, bufP);

    // ---- SA1 MLP: 32->64 (Q), 64->64 (P), 64->128 (Q) ----
    zero_f32<<<8, 256, 0, stream>>>(cStats, 2048);
    gemm_bn_wmma<<<(int)(M1 / 256), 256, 0, stream>>>(bufP, 32, Wb[0], 32, bp[0],
        nullptr, nullptr, bufQ, 64, cSum, cSqs, 64, 32);
    finalize_bn<<<1, 256, 0, stream>>>(cSum, cSqs, gp[0], ep[0], scaleL[0], shiftL[0], 64, invM1);

    zero_f32<<<8, 256, 0, stream>>>(cStats, 2048);
    gemm_bn_wmma<<<(int)(M1 / 256), 256, 0, stream>>>(bufQ, 64, Wb[1], 64, bp[1],
        scaleL[0], shiftL[0], bufP, 64, cSum, cSqs, 64, 64);
    finalize_bn<<<1, 256, 0, stream>>>(cSum, cSqs, gp[1], ep[1], scaleL[1], shiftL[1], 64, invM1);

    zero_f32<<<8, 256, 0, stream>>>(cStats, 2048);
    gemm_bn_wmma<<<(int)(M1 / 256), 256, 0, stream>>>(bufP, 64, Wb[2], 64, bp[2],
        scaleL[1], shiftL[1], bufQ, 128, cSum, cSqs, 128, 64);
    finalize_bn<<<1, 256, 0, stream>>>(cSum, cSqs, gp[2], ep[2], scaleL[2], shiftL[2], 128, invM1);

    // 5) BN+ReLU + radius-masked max pool -> x1 and rt_pp_feat output
    maxpool_sa1<<<BB * S1, 128, 0, stream>>>(bufQ, scaleL[2], shiftL[2], topkd2, x1, out_rt);

    // 6) SA2 input
    build_g2<<<(int)((M2 * 160 + 255) / 256), 256, 0, stream>>>(sampled, x1, g2);

    // ---- SA2 MLP: 160->128, 128->128, 128->1024 ----
    zero_f32<<<8, 256, 0, stream>>>(cStats, 2048);
    gemm_bn_wmma<<<(int)(M2 / 256), 256, 0, stream>>>(g2, 160, Wb[3], 160, bp[3],
        nullptr, nullptr, Y4, 128, cSum, cSqs, 128, 160);
    finalize_bn<<<1, 256, 0, stream>>>(cSum, cSqs, gp[3], ep[3], scaleL[3], shiftL[3], 128, invM2);

    zero_f32<<<8, 256, 0, stream>>>(cStats, 2048);
    gemm_bn_wmma<<<(int)(M2 / 256), 256, 0, stream>>>(Y4, 128, Wb[4], 128, bp[4],
        scaleL[3], shiftL[3], Y5, 128, cSum, cSqs, 128, 128);
    finalize_bn<<<1, 256, 0, stream>>>(cSum, cSqs, gp[4], ep[4], scaleL[4], shiftL[4], 128, invM2);

    zero_f32<<<8, 256, 0, stream>>>(cStats, 2048);
    gemm_bn_wmma<<<(int)(M2 / 256), 256, 0, stream>>>(Y5, 128, Wb[5], 128, bp[5],
        scaleL[4], shiftL[4], Y6, 1024, cSum, cSqs, 1024, 128);
    finalize_bn<<<4, 256, 0, stream>>>(cSum, cSqs, gp[5], ep[5], scaleL[5], shiftL[5], 1024, invM2);

    // 7) global max pool -> global_x
    global_max<<<(BB * 1024 + 255) / 256, 256, 0, stream>>>(Y6, scaleL[5], shiftL[5], out_gx);
}
